// ConvMatchLayerBlock_57552561766858
// MI455X (gfx1250) — matrix operand order repaired and checked by reading, not compiled, not run on hardware
//
#include <hip/hip_runtime.h>
#include <hip/hip_bf16.h>
#include <math.h>

// ---------------------------------------------------------------------------
// Problem constants (from reference): C=128, HEAD=4, GRID=16, B=32, N=2000
// ---------------------------------------------------------------------------
#define BATCH 32
#define CH    128
#define CH2   256
#define NPTS  2000
#define GL    256      // GRID*GRID
#define NH    4
#define HD    32
#define K33   1152     // 128*9 (im2col K for 3x3 conv)

#define USE_ASYNC_LDS 1

typedef _Float16 half_t;
typedef __attribute__((ext_vector_type(16))) _Float16 v16h;
typedef __attribute__((ext_vector_type(8)))  _Float16 h8;
typedef __attribute__((ext_vector_type(8)))  float    v8f;

union V16 { v16h v; h8 h[2]; };

// ---------------------------------------------------------------------------
// Generic batched WMMA GEMM:  C[z] = alpha * op(A[z]) * op(B[z]) (+bias)(+res)
// op(A): M x K, op(B): K x N. f16 in, f32 accumulate, v_wmma_f32_16x16x32_f16.
// Tile 64x64, 4 waves (wave32), K-step 32, double-buffered LDS pipeline:
//   - A tile LDS-resident row-major [m][k]; B tile TRANSPOSED [n][k] so both
//     fragment loads are 2x ds_load_b128 per 16x16 fragment (bank-conflict
//     free: row stride 80B).
//   - contiguous staging paths (TA==0 for A, TB==1 for B) use
//     global_load_async_to_lds_b128 (ASYNCcnt) issued for the NEXT K tile
//     while WMMAs run on the current one; transposed staging uses b128
//     global loads + LDS scatter. Ragged edges: scalar fallback.
// ---------------------------------------------------------------------------
template<int TA, int TB>
__global__ __launch_bounds__(128) void k_wmma_gemm(
    const half_t* __restrict__ A, int lda, long sA,
    const half_t* __restrict__ Bm, int ldb, long sB,
    float* __restrict__ Cf, int ldc, long sC,
    half_t* __restrict__ Ch, int ldch, long sCh,
    const float* __restrict__ bias,
    const float* __restrict__ res,
    float alpha, int relu,
    int M, int N, int K)
{
  __shared__ half_t As[2][64][40];   // [buf][m][k], row stride 80B
  __shared__ half_t Bt[2][64][40];   // [buf][n][k], row stride 80B

  const int bz = blockIdx.z;
  A  += (long)bz * sA;
  Bm += (long)bz * sB;
  const long coff  = (long)bz * sC;
  const long choff = (long)bz * sCh;

  const int m0 = blockIdx.y * 64;
  const int n0 = blockIdx.x * 64;
  const int t    = threadIdx.x;
  const int lane = t & 31;
  const int w    = t >> 5;
  const int wm = (w >> 1) * 32;
  const int wn = (w & 1)  * 32;

  const bool alnA = ((lda & 7) == 0);
  const bool alnB = ((ldb & 7) == 0);
  const bool fullM = (m0 + 64 <= M);
  const bool fullN = (n0 + 64 <= N);

  // ---- tile staging (issues loads; completion via ASYNCcnt/DScnt + barrier)
  auto stageA = [&](int k0, int buf) {
    const bool fullK = (k0 + 32 <= K);
    if (TA == 0 && alnA && fullM && fullK) {
      #pragma unroll
      for (int i = 0; i < 2; ++i) {
        int c = i * 128 + t;                  // 256 chunks of 8 halves
        int row = c >> 2, colb = (c & 3) * 8;
        const half_t* gp = A + (long)(m0 + row) * lda + (k0 + colb);
#if USE_ASYNC_LDS
        unsigned ldsa = (unsigned)(uintptr_t)&As[buf][row][colb];
        asm volatile("global_load_async_to_lds_b128 %0, %1, off"
                     :: "v"(ldsa), "v"((unsigned long long)(uintptr_t)gp)
                     : "memory");
#else
        *(h8*)&As[buf][row][colb] = *(const h8*)gp;
#endif
      }
    } else if (TA == 1 && alnA && fullM && fullK) {
      #pragma unroll
      for (int i = 0; i < 2; ++i) {
        int c = i * 128 + t;                  // A stored K x M, contiguous in m
        int kk = c >> 3, mb = (c & 7) * 8;
        h8 v = *(const h8*)(A + (long)(k0 + kk) * lda + (m0 + mb));
        #pragma unroll
        for (int j = 0; j < 8; ++j) As[buf][mb + j][kk] = v[j];
      }
    } else {
      #pragma unroll
      for (int i = 0; i < 16; ++i) {
        int idx = i * 128 + t;
        int r = idx >> 5, c = idx & 31;
        int gm = m0 + r, gk = k0 + c;
        half_t v = (half_t)0.0f;
        if (gm < M && gk < K)
          v = TA ? A[(long)gk * lda + gm] : A[(long)gm * lda + gk];
        As[buf][r][c] = v;
      }
    }
  };

  auto stageB = [&](int k0, int buf) {
    const bool fullK = (k0 + 32 <= K);
    if (TB == 1 && alnB && fullN && fullK) {
      // B stored N x K: contiguous in k -> direct async copy into Bt[n][k]
      #pragma unroll
      for (int i = 0; i < 2; ++i) {
        int c = i * 128 + t;
        int nn = c >> 2, kb = (c & 3) * 8;
        const half_t* gp = Bm + (long)(n0 + nn) * ldb + (k0 + kb);
#if USE_ASYNC_LDS
        unsigned ldsa = (unsigned)(uintptr_t)&Bt[buf][nn][kb];
        asm volatile("global_load_async_to_lds_b128 %0, %1, off"
                     :: "v"(ldsa), "v"((unsigned long long)(uintptr_t)gp)
                     : "memory");
#else
        *(h8*)&Bt[buf][nn][kb] = *(const h8*)gp;
#endif
      }
    } else if (TB == 0 && alnB && fullN && fullK) {
      // B stored K x N: vector load along n, scatter-transpose into Bt[n][k]
      #pragma unroll
      for (int i = 0; i < 2; ++i) {
        int c = i * 128 + t;
        int kk = c >> 3, nb = (c & 7) * 8;
        h8 v = *(const h8*)(Bm + (long)(k0 + kk) * ldb + (n0 + nb));
        #pragma unroll
        for (int j = 0; j < 8; ++j) Bt[buf][nb + j][kk] = v[j];
      }
    } else {
      #pragma unroll
      for (int i = 0; i < 16; ++i) {
        int idx = i * 128 + t;
        int r = idx >> 6, c = idx & 63;   // r = k, c = n
        int gk = k0 + r, gn = n0 + c;
        half_t v = (half_t)0.0f;
        if (gk < K && gn < N)
          v = TB ? Bm[(long)gn * ldb + gk] : Bm[(long)gk * ldb + gn];
        Bt[buf][c][r] = v;
      }
    }
  };

  v8f acc[2][2] = {};

  // ---- software pipeline: stage k0=0, then overlap next-tile staging ----
  stageA(0, 0);
  stageB(0, 0);
  int cur = 0;

  for (int k0 = 0; k0 < K; k0 += 32) {
#if USE_ASYNC_LDS
    asm volatile("s_wait_asynccnt 0x0" ::: "memory");
#endif
    __syncthreads();   // buf[cur] staged & visible; prior reads of buf[cur^1] done

    if (k0 + 32 < K) {
      stageA(k0 + 32, cur ^ 1);
      stageB(k0 + 32, cur ^ 1);
    }

    // ---- fragments per CDNA5 16-bit WMMA layouts (all b128 LDS loads) ----
    const int arow = wm + (lane & 15);
    const int kb   = (lane >> 4) * 8;     // A frag: halves j<8 -> K=kb+j, j>=8 -> 16+kb+j-8
    const int bcol = wn + (lane & 15);
    const int kb2  = (lane >> 4) * 8;     // B frag: same K order as A frag

    v16h afr[2], bfr[2];
    #pragma unroll
    for (int mi = 0; mi < 2; ++mi) {
      V16 a;
      a.h[0] = *(const h8*)&As[cur][arow + mi * 16][kb];
      a.h[1] = *(const h8*)&As[cur][arow + mi * 16][16 + kb];
      afr[mi] = a.v;
    }
    #pragma unroll
    for (int ni = 0; ni < 2; ++ni) {
      V16 b;
      b.h[0] = *(const h8*)&Bt[cur][bcol + ni * 16][kb2];
      b.h[1] = *(const h8*)&Bt[cur][bcol + ni * 16][16 + kb2];
      bfr[ni] = b.v;
    }

    #pragma unroll
    for (int mi = 0; mi < 2; ++mi)
      #pragma unroll
      for (int ni = 0; ni < 2; ++ni)
        acc[mi][ni] = __builtin_amdgcn_wmma_f32_16x16x32_f16(
            false, afr[mi], false, bfr[ni], (short)0, acc[mi][ni], false, false);

    cur ^= 1;
  }

  // ---- epilogue (C layout: VGPR r -> M = r + 8*(lane>=16), N = lane&15) ----
  #pragma unroll
  for (int mi = 0; mi < 2; ++mi) {
    #pragma unroll
    for (int ni = 0; ni < 2; ++ni) {
      int rbase = m0 + wm + mi * 16 + (lane >> 4) * 8;
      int col   = n0 + wn + ni * 16 + (lane & 15);
      if (col >= N) continue;
      #pragma unroll
      for (int r = 0; r < 8; ++r) {
        int row = rbase + r;
        if (row >= M) continue;
        float v = alpha * acc[mi][ni][r];
        if (bias) v += bias[row];
        if (res)  v += res[coff + (long)row * ldc + col];
        if (relu) v = fmaxf(v, 0.f);
        if (Cf) Cf[coff  + (long)row * ldc  + col] = v;
        if (Ch) Ch[choff + (long)row * ldch + col] = (half_t)v;
      }
    }
  }
}

// ---------------------------------------------------------------------------
// Elementwise / reduction helpers
// ---------------------------------------------------------------------------
__global__ void k_cast_h(const float* __restrict__ s, half_t* __restrict__ d, long n) {
  long stride = (long)gridDim.x * blockDim.x;
  for (long i = (long)blockIdx.x * blockDim.x + threadIdx.x; i < n; i += stride)
    d[i] = (half_t)s[i];
}

__global__ void k_sub_h(const float* __restrict__ a, const float* __restrict__ b,
                        half_t* __restrict__ d, long n) {
  long stride = (long)gridDim.x * blockDim.x;
  for (long i = (long)blockIdx.x * blockDim.x + threadIdx.x; i < n; i += stride)
    d[i] = (half_t)(a[i] - b[i]);
}

// pack f32 (B,Cc,L) into f16 (B,Cdst,L) at channel offset coff (for concat)
__global__ void k_pack_h(const float* __restrict__ s, half_t* __restrict__ d,
                         int Cc, int L, int Cdst, int coff, long total) {
  long stride = (long)gridDim.x * blockDim.x;
  for (long i = (long)blockIdx.x * blockDim.x + threadIdx.x; i < total; i += stride) {
    long l = i % L;
    long cc = (i / L) % Cc;
    long b = i / ((long)Cc * L);
    d[(b * Cdst + coff + cc) * L + l] = (half_t)s[i];
  }
}

// in-place softmax over rows of f16 (one block per row)
__global__ void k_softmax_h(half_t* __restrict__ s, int L) {
  long row = blockIdx.x;
  half_t* p = s + row * (long)L;
  __shared__ float red[256];
  int t = threadIdx.x;
  float mx = -3.0e38f;
  for (int i = t; i < L; i += 256) mx = fmaxf(mx, (float)p[i]);
  red[t] = mx; __syncthreads();
  for (int o = 128; o > 0; o >>= 1) { if (t < o) red[t] = fmaxf(red[t], red[t + o]); __syncthreads(); }
  mx = red[0]; __syncthreads();
  float sum = 0.f;
  for (int i = t; i < L; i += 256) { float e = expf((float)p[i] - mx); p[i] = (half_t)e; sum += e; }
  red[t] = sum; __syncthreads();
  for (int o = 128; o > 0; o >>= 1) { if (t < o) red[t] += red[t + o]; __syncthreads(); }
  float inv = 1.f / red[0];
  __syncthreads();
  for (int i = t; i < L; i += 256) p[i] = (half_t)((float)p[i] * inv);
}

// batch-norm statistics over axes (batch, L): one block per channel
__global__ void k_bn_stats(const float* __restrict__ x, float* __restrict__ mv,
                           int Cc, int Bn, int L) {
  int c = blockIdx.x;
  int t = threadIdx.x;
  __shared__ float s1[256], s2[256];
  float a = 0.f, b = 0.f;
  long per = (long)Bn * L;
  for (long i = t; i < per; i += 256) {
    long bb = i / L, l = i % L;
    float v = x[(bb * Cc + c) * (long)L + l];
    a += v; b += v * v;
  }
  s1[t] = a; s2[t] = b; __syncthreads();
  for (int o = 128; o > 0; o >>= 1) { if (t < o) { s1[t] += s1[t + o]; s2[t] += s2[t + o]; } __syncthreads(); }
  if (t == 0) {
    float m = s1[0] / (float)per;
    mv[2 * c] = m;
    mv[2 * c + 1] = s2[0] / (float)per - m * m;
  }
}

// y = relu?( (x-m)*rsqrt(v+eps)*g + be ) [+ res];  optional f16 shadow
__global__ void k_bn_apply(const float* __restrict__ x, const float* __restrict__ mv,
                           const float* __restrict__ g, const float* __restrict__ be,
                           float* __restrict__ y, half_t* __restrict__ yh,
                           const float* __restrict__ res,
                           int Cc, int L, long total, int relu, float eps) {
  long stride = (long)gridDim.x * blockDim.x;
  for (long i = (long)blockIdx.x * blockDim.x + threadIdx.x; i < total; i += stride) {
    int c = (int)((i / L) % Cc);
    float v = (x[i] - mv[2 * c]) * rsqrtf(mv[2 * c + 1] + eps) * g[c] + be[c];
    if (relu) v = fmaxf(v, 0.f);
    if (res)  v += res[i];
    if (y)  y[i]  = v;
    if (yh) yh[i] = (half_t)v;
  }
}

// instance norm over L (one block per (b,c) row), in place
__global__ void k_inorm(float* __restrict__ x, int L, float eps) {
  long row = blockIdx.x;
  float* p = x + row * (long)L;
  __shared__ float s1[256], s2[256];
  int t = threadIdx.x;
  float a = 0.f, b = 0.f;
  for (int i = t; i < L; i += 256) { float v = p[i]; a += v; b += v * v; }
  s1[t] = a; s2[t] = b; __syncthreads();
  for (int o = 128; o > 0; o >>= 1) { if (t < o) { s1[t] += s1[t + o]; s2[t] += s2[t + o]; } __syncthreads(); }
  float m = s1[0] / (float)L;
  float inv = rsqrtf(s2[0] / (float)L - m * m + eps);
  __syncthreads();
  for (int i = t; i < L; i += 256) p[i] = (p[i] - m) * inv;
}

// im2col for 3x3 SAME conv on (B,128,16,16) -> f16 (B,1152,256)
__global__ void k_im2col(const float* __restrict__ x, half_t* __restrict__ col, long total) {
  long stride = (long)gridDim.x * blockDim.x;
  for (long i = (long)blockIdx.x * blockDim.x + threadIdx.x; i < total; i += stride) {
    int ppos = (int)(i % 256);
    long r = i / 256;
    int j = (int)(r % 9); r /= 9;
    int c = (int)(r % CH);
    long b = r / CH;
    int yy = ppos >> 4, xx = ppos & 15;
    int sy = yy + j / 3 - 1, sx = xx + j % 3 - 1;
    float v = 0.f;
    if ((unsigned)sy < 16u && (unsigned)sx < 16u)
      v = x[(b * CH + c) * 256 + sy * 16 + sx];
    col[i] = (half_t)v;
  }
}

// XwX[b] = sum_n relu(tanh(logit)) * X[n]^T X[n]   (81 threads per batch)
__global__ void k_xwx(const float* __restrict__ xs, const float* __restrict__ logits,
                      float* __restrict__ XwX) {
  int b = blockIdx.x;
  int t = threadIdx.x;
  if (t >= 81) return;
  int p = t / 9, q = t % 9;
  const float* xb = xs + (long)b * NPTS * 4;
  const float* lg = logits + (long)b * NPTS;
  float acc = 0.f;
  for (int n = 0; n < NPTS; ++n) {
    float x0 = xb[n * 4 + 0], x1 = xb[n * 4 + 1];
    float x2 = xb[n * 4 + 2], x3 = xb[n * 4 + 3];
    float w = tanhf(lg[n]); w = fmaxf(w, 0.f);
    float X[9] = { x2 * x0, x2 * x1, x2, x3 * x0, x3 * x1, x3, x0, x1, 1.f };
    acc += w * X[p] * X[q];
  }
  XwX[(long)b * 81 + t] = acc;
}

// 9x9 symmetric Jacobi eigensolver; emit unit eigenvector of min eigenvalue
__global__ void k_eigh9(const float* __restrict__ M9, float* __restrict__ e) {
  int b = threadIdx.x;
  if (b >= BATCH) return;
  float A[9][9], V[9][9];
  for (int i = 0; i < 9; ++i)
    for (int j = 0; j < 9; ++j) {
      A[i][j] = M9[(long)b * 81 + i * 9 + j];
      V[i][j] = (i == j) ? 1.f : 0.f;
    }
  for (int sweep = 0; sweep < 40; ++sweep) {
    for (int p = 0; p < 8; ++p)
      for (int q = p + 1; q < 9; ++q) {
        float apq = A[p][q];
        if (fabsf(apq) < 1e-12f) continue;
        float theta = (A[q][q] - A[p][p]) / (2.f * apq);
        float tt = (theta >= 0.f ? 1.f : -1.f) / (fabsf(theta) + sqrtf(theta * theta + 1.f));
        float cc = rsqrtf(tt * tt + 1.f);
        float ss = tt * cc;
        for (int k = 0; k < 9; ++k) { float a1 = A[k][p], a2 = A[k][q]; A[k][p] = cc * a1 - ss * a2; A[k][q] = ss * a1 + cc * a2; }
        for (int k = 0; k < 9; ++k) { float a1 = A[p][k], a2 = A[q][k]; A[p][k] = cc * a1 - ss * a2; A[q][k] = ss * a1 + cc * a2; }
        for (int k = 0; k < 9; ++k) { float v1 = V[k][p], v2 = V[k][q]; V[k][p] = cc * v1 - ss * v2; V[k][q] = ss * v1 + cc * v2; }
      }
  }
  int mi = 0; float mv = A[0][0];
  for (int i = 1; i < 9; ++i) if (A[i][i] < mv) { mv = A[i][i]; mi = i; }
  float nrm = 0.f;
  for (int i = 0; i < 9; ++i) nrm += V[i][mi] * V[i][mi];
  nrm = rsqrtf(nrm);
  for (int i = 0; i < 9; ++i) e[(long)b * 9 + i] = V[i][mi] * nrm;
}

// ---------------------------------------------------------------------------
// Host-side helpers
// ---------------------------------------------------------------------------
static inline void gemm(hipStream_t st,
    const half_t* A, int lda, long sA, int tA,
    const half_t* Bm, int ldb, long sB, int tB,
    float* Cf, int ldc, long sC, half_t* Ch, int ldch, long sCh,
    const float* bias, const float* res, float alpha, int relu,
    int M, int N, int K, int batch) {
  dim3 g((N + 63) / 64, (M + 63) / 64, batch);
  if (tA == 0 && tB == 0)
    k_wmma_gemm<0, 0><<<g, 128, 0, st>>>(A, lda, sA, Bm, ldb, sB, Cf, ldc, sC,
                                         Ch, ldch, sCh, bias, res, alpha, relu, M, N, K);
  else if (tA == 1)
    k_wmma_gemm<1, 0><<<g, 128, 0, st>>>(A, lda, sA, Bm, ldb, sB, Cf, ldc, sC,
                                         Ch, ldch, sCh, bias, res, alpha, relu, M, N, K);
  else
    k_wmma_gemm<0, 1><<<g, 128, 0, st>>>(A, lda, sA, Bm, ldb, sB, Cf, ldc, sC,
                                         Ch, ldch, sCh, bias, res, alpha, relu, M, N, K);
}

static inline int nblk(long total) {
  long b = (total + 255) / 256;
  return (int)(b > 16384 ? 16384 : b);
}

struct AttnW {
  const half_t *wq, *wk, *wv, *wmh, *wc1, *wc2;
  const float  *bq, *bk, *bv, *bmh, *bc1, *bc2, *bg, *bbe;
};

// one _attention block: outf = m1 + cat2( relu(bn(cat1( [m1; mh(attn)] ))) )
static void attention(hipStream_t st, const AttnW& W,
    const float* m1f, const half_t* m1h, const half_t* m2h, int L1, int L2,
    float* outf,
    half_t* qh, half_t* kh, half_t* vh, half_t* sch,
    half_t* addh, half_t* cath, float* catf, float* mv) {
  // q/k/v projections (f16-only outputs)
  gemm(st, W.wq, CH, 0, 0, m1h, L1, (long)CH * L1, 0,
       nullptr, 0, 0, qh, L1, (long)CH * L1, W.bq, nullptr, 1.f, 0, CH, L1, CH, BATCH);
  gemm(st, W.wk, CH, 0, 0, m2h, L2, (long)CH * L2, 0,
       nullptr, 0, 0, kh, L2, (long)CH * L2, W.bk, nullptr, 1.f, 0, CH, L2, CH, BATCH);
  gemm(st, W.wv, CH, 0, 0, m2h, L2, (long)CH * L2, 0,
       nullptr, 0, 0, vh, L2, (long)CH * L2, W.bv, nullptr, 1.f, 0, CH, L2, CH, BATCH);
  // score = q^T k / sqrt(HD)   (per (b,h); f16 only)
  gemm(st, qh, L1, (long)HD * L1, 1, kh, L2, (long)HD * L2, 0,
       nullptr, 0, 0, sch, L2, (long)L1 * L2, nullptr, nullptr,
       0.17677669529663688f, 0, L1, L2, HD, BATCH * NH);
  k_softmax_h<<<dim3((unsigned)(BATCH * NH * L1)), 256, 0, st>>>(sch, L2);
  // add = v * score^T
  gemm(st, vh, L2, (long)HD * L2, 0, sch, L2, (long)L1 * L2, 1,
       nullptr, 0, 0, addh, L1, (long)HD * L1, nullptr, nullptr, 1.f, 0, HD, L1, L2, BATCH * NH);
  // mh projection straight into concat buffer channels [C, 2C)
  gemm(st, W.wmh, CH, 0, 0, addh, L1, (long)CH * L1, 0,
       nullptr, 0, 0, cath + (long)CH * L1, L1, (long)CH2 * L1,
       W.bmh, nullptr, 1.f, 0, CH, L1, CH, BATCH);
  // concat channels [0, C) = m1
  {
    long tot = (long)BATCH * CH * L1;
    k_pack_h<<<nblk(tot), 256, 0, st>>>(m1f, cath, CH, L1, CH2, 0, tot);
  }
  // cat1 (2C x 2C)
  gemm(st, W.wc1, CH2, 0, 0, cath, L1, (long)CH2 * L1, 0,
       catf, L1, (long)CH2 * L1, nullptr, 0, 0, W.bc1, nullptr, 1.f, 0, CH2, L1, CH2, BATCH);
  // bn + relu (f16 shadow reuses cath — cat1 already consumed it)
  k_bn_stats<<<dim3(CH2), 256, 0, st>>>(catf, mv, CH2, BATCH, L1);
  {
    long tot = (long)BATCH * CH2 * L1;
    k_bn_apply<<<nblk(tot), 256, 0, st>>>(catf, mv, W.bg, W.bbe, catf, cath,
                                          nullptr, CH2, L1, tot, 1, 1e-5f);
  }
  // cat2 (C x 2C) + residual m1
  gemm(st, W.wc2, CH2, 0, 0, cath, L1, (long)CH2 * L1, 0,
       outf, L1, (long)CH * L1, nullptr, 0, 0, W.bc2, m1f, 1.f, 0, CH, L1, CH2, BATCH);
}

// ---------------------------------------------------------------------------
// kernel_launch
// ---------------------------------------------------------------------------
extern "C" void kernel_launch(void* const* d_in, const int* in_sizes, int n_in,
                              void* d_out, int out_size, void* d_ws, size_t ws_size,
                              hipStream_t stream) {
  (void)in_sizes; (void)n_in; (void)out_size; (void)ws_size;

  // ---- input indexing (flattened param dict, insertion order) ----
  const float* xs  = (const float*)d_in[0];
  const float* din = (const float*)d_in[1];
  const float* gpe = (const float*)d_in[2];
  const int ALIGN = 3, DEALIGN = 17, FILT = 31, SCALE_W = 55, INL = 57;
  auto P = [&](int i) { return (const float*)d_in[i]; };

  float* out = (float*)d_out;
  const long dnew_off = 0;
  const long logit_off = (long)BATCH * CH * NPTS;               // 8,192,000
  const long ehat_off  = logit_off + (long)BATCH * NPTS;        // 8,256,000
  const long gridd_off = ehat_off + (long)BATCH * 9;            // 8,256,288

  // ---- workspace bump allocator ----
  char* base = (char*)d_ws;
  size_t cur = 0;
  auto alloc = [&](size_t bytes) -> void* {
    void* p = base + cur;
    cur += (bytes + 255) & ~(size_t)255;
    return p;
  };
  const long BCN = (long)BATCH * CH * NPTS;
  const long BCG = (long)BATCH * CH * GL;

  half_t* dh     = (half_t*)alloc(BCN * 2);
  half_t* gpeh   = (half_t*)alloc(BCG * 2);
  half_t* qh     = (half_t*)alloc(BCN * 2);
  half_t* kh     = (half_t*)alloc(BCN * 2);
  half_t* vh     = (half_t*)alloc(BCN * 2);
  half_t* sch    = (half_t*)alloc((long)BATCH * NH * NPTS * GL * 2);
  half_t* addh   = (half_t*)alloc(BCN * 2);
  half_t* cath   = (half_t*)alloc((long)BATCH * CH2 * NPTS * 2);
  float*  catf   = (float*)alloc((long)BATCH * CH2 * NPTS * 4);
  float*  gridd0 = (float*)alloc(BCG * 4);
  float*  gridd1 = (float*)alloc(BCG * 4);
  half_t* gxh    = (half_t*)alloc(BCG * 2);
  half_t* gridh  = (half_t*)alloc(BCG * 2);
  half_t* colh   = (half_t*)alloc((long)BATCH * K33 * GL * 2);
  float*  tbuf   = (float*)alloc(BCG * 4);
  float*  mv     = (float*)alloc(2 * CH2 * 4);
  float*  it0    = (float*)alloc((long)BATCH * 64 * NPTS * 4);
  half_t* it0h   = (half_t*)alloc((long)BATCH * 64 * NPTS * 2);
  float*  it1    = (float*)alloc((long)BATCH * 16 * NPTS * 4);
  half_t* it1h   = (half_t*)alloc((long)BATCH * 16 * NPTS * 2);
  float*  it2    = (float*)alloc((long)BATCH * 4 * NPTS * 4);
  half_t* it2h   = (half_t*)alloc((long)BATCH * 4 * NPTS * 2);
  half_t* diffh  = (half_t*)alloc(BCN * 2);
  float*  xwx    = (float*)alloc((long)BATCH * 81 * 4);
  half_t* wpool  = (half_t*)alloc(4 * 1024 * 1024);   // f16 weights

  // ---- cast weights to f16 ----
  size_t wcur = 0;
  auto castw = [&](const float* src, long n) -> half_t* {
    half_t* dst = wpool + wcur;
    wcur += (size_t)((n + 127) & ~127L);
    k_cast_h<<<nblk(n), 256, 0, stream>>>(src, dst, n);
    return dst;
  };
  auto mk_attn = [&](int pb) {
    AttnW W;
    W.wq  = castw(P(pb + 0),  CH * CH);   W.bq  = P(pb + 1);
    W.wk  = castw(P(pb + 2),  CH * CH);   W.bk  = P(pb + 3);
    W.wv  = castw(P(pb + 4),  CH * CH);   W.bv  = P(pb + 5);
    W.wmh = castw(P(pb + 6),  CH * CH);   W.bmh = P(pb + 7);
    W.wc1 = castw(P(pb + 8),  CH2 * CH2); W.bc1 = P(pb + 9);
    W.bg  = P(pb + 10);                   W.bbe = P(pb + 11);
    W.wc2 = castw(P(pb + 12), CH * CH2);  W.bc2 = P(pb + 13);
    return W;
  };
  AttnW Wa = mk_attn(ALIGN);
  AttnW Wd = mk_attn(DEALIGN);
  half_t* fw1[3]; half_t* fw2[3];
  for (int rb = 0; rb < 3; ++rb) {
    fw1[rb] = castw(P(FILT + rb * 8 + 0), CH * K33);
    fw2[rb] = castw(P(FILT + rb * 8 + 4), CH * K33);
  }
  half_t* scaleh = castw(P(SCALE_W), CH * CH);
  half_t* ic0 = castw(P(INL + 0), 64 * CH);
  half_t* ic1 = castw(P(INL + 2), 16 * 64);
  half_t* ic2 = castw(P(INL + 4), 4 * 16);
  half_t* ic3 = castw(P(INL + 6), 1 * 4);

  // ---- cast activations ----
  k_cast_h<<<nblk(BCN), 256, 0, stream>>>(din, dh, BCN);
  k_cast_h<<<nblk(BCG), 256, 0, stream>>>(gpe, gpeh, BCG);

  // ---- 1) align attention: grid_d0 = attn(gpe, d) ----
  attention(stream, Wa, gpe, gpeh, dh, GL, NPTS, gridd0,
            qh, kh, vh, sch, addh, cath, catf, mv);

  // ---- 2) filter (3 res blocks of 3x3 conv + global BN + relu, then scale) ----
  float* xcur = gridd0;
  float* xnxt = gridd1;
  const long col_tot = (long)BATCH * K33 * GL;
  const long bcg_tot = BCG;
  for (int rb = 0; rb < 3; ++rb) {
    const float* b1  = P(FILT + rb * 8 + 1);
    const float* g1  = P(FILT + rb * 8 + 2);
    const float* be1 = P(FILT + rb * 8 + 3);
    const float* b2  = P(FILT + rb * 8 + 5);
    const float* g2  = P(FILT + rb * 8 + 6);
    const float* be2 = P(FILT + rb * 8 + 7);

    k_im2col<<<nblk(col_tot), 256, 0, stream>>>(xcur, colh, col_tot);
    gemm(stream, fw1[rb], K33, 0, 0, colh, GL, (long)K33 * GL, 0,
         tbuf, GL, (long)CH * GL, nullptr, 0, 0, b1, nullptr, 1.f, 0, CH, GL, K33, BATCH);
    k_bn_stats<<<dim3(CH), 256, 0, stream>>>(tbuf, mv, CH, BATCH, GL);
    k_bn_apply<<<nblk(bcg_tot), 256, 0, stream>>>(tbuf, mv, g1, be1, tbuf, nullptr,
                                                  nullptr, CH, GL, bcg_tot, 1, 1e-5f);
    k_im2col<<<nblk(col_tot), 256, 0, stream>>>(tbuf, colh, col_tot);
    gemm(stream, fw2[rb], K33, 0, 0, colh, GL, (long)K33 * GL, 0,
         tbuf, GL, (long)CH * GL, nullptr, 0, 0, b2, nullptr, 1.f, 0, CH, GL, K33, BATCH);
    k_bn_stats<<<dim3(CH), 256, 0, stream>>>(tbuf, mv, CH, BATCH, GL);
    // relu(bn(.)) + residual; f16 shadow gxh includes the residual
    k_bn_apply<<<nblk(bcg_tot), 256, 0, stream>>>(tbuf, mv, g2, be2, xnxt, gxh,
                                                  xcur, CH, GL, bcg_tot, 1, 1e-5f);
    float* tmp = xcur; xcur = xnxt; xnxt = tmp;
  }
  gemm(stream, scaleh, CH, 0, 0, gxh, GL, (long)CH * GL, 0,
       out + gridd_off, GL, (long)CH * GL, gridh, GL, (long)CH * GL,
       P(SCALE_W + 1), nullptr, 1.f, 0, CH, GL, CH, BATCH);

  // ---- 3) dealign attention: d_new = attn(d, grid_d) ----
  attention(stream, Wd, din, dh, gridh, NPTS, GL, out + dnew_off,
            qh, kh, vh, sch, addh, cath, catf, mv);

  // ---- 4) inlier predictor on (d_new - d) ----
  k_sub_h<<<nblk(BCN), 256, 0, stream>>>(out + dnew_off, din, diffh, BCN);

  gemm(stream, ic0, CH, 0, 0, diffh, NPTS, (long)CH * NPTS, 0,
       it0, NPTS, (long)64 * NPTS, nullptr, 0, 0, P(INL + 1), nullptr, 1.f, 0, 64, NPTS, CH, BATCH);
  k_inorm<<<dim3(BATCH * 64), 256, 0, stream>>>(it0, NPTS, 1e-3f);
  k_bn_stats<<<dim3(64), 256, 0, stream>>>(it0, mv, 64, BATCH, NPTS);
  {
    long tot = (long)BATCH * 64 * NPTS;
    k_bn_apply<<<nblk(tot), 256, 0, stream>>>(it0, mv, P(INL + 8), P(INL + 9),
                                              it0, it0h, nullptr, 64, NPTS, tot, 1, 1e-5f);
  }
  gemm(stream, ic1, 64, 0, 0, it0h, NPTS, (long)64 * NPTS, 0,
       it1, NPTS, (long)16 * NPTS, nullptr, 0, 0, P(INL + 3), nullptr, 1.f, 0, 16, NPTS, 64, BATCH);
  k_inorm<<<dim3(BATCH * 16), 256, 0, stream>>>(it1, NPTS, 1e-3f);
  k_bn_stats<<<dim3(16), 256, 0, stream>>>(it1, mv, 16, BATCH, NPTS);
  {
    long tot = (long)BATCH * 16 * NPTS;
    k_bn_apply<<<nblk(tot), 256, 0, stream>>>(it1, mv, P(INL + 10), P(INL + 11),
                                              it1, it1h, nullptr, 16, NPTS, tot, 1, 1e-5f);
  }
  gemm(stream, ic2, 16, 0, 0, it1h, NPTS, (long)16 * NPTS, 0,
       it2, NPTS, (long)4 * NPTS, nullptr, 0, 0, P(INL + 5), nullptr, 1.f, 0, 4, NPTS, 16, BATCH);
  k_inorm<<<dim3(BATCH * 4), 256, 0, stream>>>(it2, NPTS, 1e-3f);
  k_bn_stats<<<dim3(4), 256, 0, stream>>>(it2, mv, 4, BATCH, NPTS);
  {
    long tot = (long)BATCH * 4 * NPTS;
    k_bn_apply<<<nblk(tot), 256, 0, stream>>>(it2, mv, P(INL + 12), P(INL + 13),
                                              it2, it2h, nullptr, 4, NPTS, tot, 1, 1e-5f);
  }
  gemm(stream, ic3, 4, 0, 0, it2h, NPTS, (long)4 * NPTS, 0,
       out + logit_off, NPTS, (long)NPTS, nullptr, 0, 0, P(INL + 7), nullptr,
       1.f, 0, 1, NPTS, 4, BATCH);

  // ---- 5) weighted 8-point: XwX then 9x9 eigh ----
  k_xwx<<<dim3(BATCH), 128, 0, stream>>>(xs, out + logit_off, xwx);
  k_eigh9<<<dim3(1), 32, 0, stream>>>(xwx, out + ehat_off);
}
